// HybridQuantumRegressionModel_65481071409932
// MI455X (gfx1250) — matrix operand order repaired and checked by reading, compiled, not run on hardware
//
#include <hip/hip_runtime.h>

// ---------------------------------------------------------------------------
// out = X(65536x512) @ coef(512) + const      (whole reference folds to GEMV)
// Memory-bound: 128 MB stream -> ~5.5 us at 23.3 TB/s HBM (fits in 192MB L2).
// Matrix path: v_wmma_f32_16x16x32_f16 with hi/lo f16 error-compensated split
// so the result is f32-accurate while using the CDNA5 WMMA pipe.
// ---------------------------------------------------------------------------

#define NW    512
#define BATCHN 65536

typedef __attribute__((ext_vector_type(16))) _Float16 v16h;
typedef __attribute__((ext_vector_type(8)))  float    v8f;
typedef __attribute__((ext_vector_type(4)))  float    f32x4;

// ---- prep: fold all per-wire constants into coef (split hi/lo f16) + const ----
__global__ __launch_bounds__(NW) void hqr_prep_kernel(
    const float* __restrict__ sq,  const float* __restrict__ ph,
    const float* __restrict__ dp,  const float* __restrict__ sq2,
    const float* __restrict__ hw,  const float* __restrict__ hb,
    _Float16* __restrict__ cHi, _Float16* __restrict__ cLo,
    float* __restrict__ cst)
{
    __shared__ float red[NW];
    const int w = threadIdx.x;

    const float e2   = expf(-sq2[w]);                      // exp(-squeeze2)
    const float a    = e2 * 2.0f * expf(-sq[w]) * cosf(ph[w]); // sqrt(2*hbar)=2
    const float coef = a * hw[w];

    const _Float16 h = (_Float16)coef;                     // hi/lo split of coef
    cHi[w] = h;
    cLo[w] = (_Float16)(coef - (float)h);

    red[w] = e2 * 2.0f * dp[w] * hw[w];                    // constant term piece
    __syncthreads();
    for (int s = NW / 2; s > 0; s >>= 1) {
        if (w < s) red[w] += red[w + s];
        __syncthreads();
    }
    if (w == 0) *cst = red[0] + hb[0];
}

// ---- main GEMV: one wave = 16 batch rows, K=512 in 16 tiles of 32 ----
// A layout (16-bit A 16x32, wave32): lane = (hi,r); v16h elems 0..15 = K kb+16*hi..+15
// B (32x16) broadcast coef across all N: lane elems e = coef[kb + 16*hi + e]
// C/D (16x16 f32, 8 VGPRs): lane N==0 holds rows rowbase+8*hi+0..7 -> store those.
__global__ __launch_bounds__(256) void hqr_gemv_wmma_kernel(
    const float* __restrict__ X,
    const _Float16* __restrict__ cHi, const _Float16* __restrict__ cLo,
    const float* __restrict__ cstP, float* __restrict__ out)
{
    const int lane    = threadIdx.x & 31;
    const int wave    = blockIdx.x * 8 + (threadIdx.x >> 5);
    const int rowbase = wave * 16;
    const int hi      = lane >> 4;
    const int r       = lane & 15;

    const float* rowp = X + (size_t)(rowbase + r) * NW;

    v8f acc = {};
    #pragma unroll 4
    for (int t = 0; t < 16; ++t) {
        const int kb = t * 32;

        // one contiguous 16-float chunk per lane (coalesced b128 loads)
        f32x4 c0 = *(const f32x4*)(rowp + kb + 16 * hi);
        f32x4 c1 = *(const f32x4*)(rowp + kb + 16 * hi + 4);
        f32x4 c2 = *(const f32x4*)(rowp + kb + 16 * hi + 8);
        f32x4 c3 = *(const f32x4*)(rowp + kb + 16 * hi + 12);

        float xf[16];
        #pragma unroll
        for (int i = 0; i < 4; ++i) {
            xf[i] = c0[i]; xf[4 + i] = c1[i]; xf[8 + i] = c2[i]; xf[12 + i] = c3[i];
        }

        v16h ah, al;                                       // hi/lo split of x
        #pragma unroll
        for (int i = 0; i < 16; ++i) {
            const _Float16 h = (_Float16)xf[i];
            ah[i] = h;
            al[i] = (_Float16)(xf[i] - (float)h);
        }

        // coef broadcast across N (all lanes in half-wave read same 32B)
        const v16h bh = *(const v16h*)(cHi + kb + 16 * hi);
        const v16h bl = *(const v16h*)(cLo + kb + 16 * hi);

        // D = A*B + C, accumulate hi*hi + lo*hi + hi*lo (f32-accurate)
        acc = __builtin_amdgcn_wmma_f32_16x16x32_f16(false, ah, false, bh,
                                                     (short)0, acc, false, false);
        acc = __builtin_amdgcn_wmma_f32_16x16x32_f16(false, al, false, bh,
                                                     (short)0, acc, false, false);
        acc = __builtin_amdgcn_wmma_f32_16x16x32_f16(false, ah, false, bl,
                                                     (short)0, acc, false, false);
    }

    const float cst = *cstP;
    if (r == 0) {   // lanes 0 and 16: column N=0 of D holds the 16 row results
        f32x4 o0 = { acc[0] + cst, acc[1] + cst, acc[2] + cst, acc[3] + cst };
        f32x4 o1 = { acc[4] + cst, acc[5] + cst, acc[6] + cst, acc[7] + cst };
        *(f32x4*)(out + rowbase + 8 * hi)     = o0;
        *(f32x4*)(out + rowbase + 8 * hi + 4) = o1;
    }
}

extern "C" void kernel_launch(void* const* d_in, const int* in_sizes, int n_in,
                              void* d_out, int out_size, void* d_ws, size_t ws_size,
                              hipStream_t stream) {
    (void)in_sizes; (void)n_in; (void)out_size; (void)ws_size;
    // setup_inputs order: state_batch, squeeze, phase, disp, squeeze2, kerr, head_w, head_b
    const float* X   = (const float*)d_in[0];
    const float* sq  = (const float*)d_in[1];
    const float* ph  = (const float*)d_in[2];
    const float* dp  = (const float*)d_in[3];
    const float* sq2 = (const float*)d_in[4];
    /* kerr d_in[5] unused by reference */
    const float* hw  = (const float*)d_in[6];
    const float* hb  = (const float*)d_in[7];
    float* out = (float*)d_out;

    _Float16* cHi = (_Float16*)d_ws;                        // 512 halves (1 KB)
    _Float16* cLo = cHi + NW;                               // 512 halves (1 KB)
    float*    cst = (float*)((char*)d_ws + 2048);           // folded constant

    hqr_prep_kernel<<<1, NW, 0, stream>>>(sq, ph, dp, sq2, hw, hb, cHi, cLo, cst);

    // 65536 rows / (16 rows/wave * 8 waves/block) = 512 blocks
    hqr_gemv_wmma_kernel<<<BATCHN / (16 * 8), 256, 0, stream>>>(X, cHi, cLo, cst, out);
}
